// SSTGroundR_81819126989495
// MI455X (gfx1250) — matrix operand order repaired and checked by reading, not compiled, not run on hardware
//
#include <hip/hip_runtime.h>
#include <hip/hip_bf16.h>
#include <stdint.h>

// ---------------- WMMA types & helpers (gfx1250, wave32) ----------------
typedef __attribute__((ext_vector_type(16))) __bf16 bf16x16;
typedef __attribute__((ext_vector_type(8)))  float  f32x8;

__device__ __forceinline__ f32x8 wmma_bf16(bf16x16 a, bf16x16 b, f32x8 c) {
  // D = A(16x32) x B(32x16) + C, f32 accumulate
  return __builtin_amdgcn_wmma_f32_16x16x32_bf16(false, a, false, b, (short)0, c,
                                                 false, false);
}

// ---- async global->LDS staging of one 16x32 f32 A tile (2 KB) ----
// 128 threads: thread tid copies 16 bytes; row = tid/8, 4-float chunk q = tid%8.
// VDST operand = per-lane LDS byte address (low 32 bits of generic LDS pointer).
__device__ __forceinline__ void stage_a_async(float* tile /* [16][32] in LDS */,
                                              const float* __restrict__ A, long ld,
                                              int m0, int k0, int tid) {
  const int row = tid >> 3;
  const int q   = tid & 7;
  const float* g = A + (long)(m0 + row) * ld + k0 + q * 4;
  uint32_t lds_addr = (uint32_t)(uintptr_t)(tile + row * 32 + q * 4);
  asm volatile("global_load_async_to_lds_b128 %0, %1, off"
               :
               : "v"(lds_addr), "v"(g)
               : "memory");
}

__device__ __forceinline__ void wait_async_le1() {
  asm volatile("s_wait_asynccnt 0x1" ::: "memory");
}
__device__ __forceinline__ void wait_async_le0() {
  asm volatile("s_wait_asynccnt 0x0" ::: "memory");
}

// A fragment from an LDS-resident 16x32 f32 tile, converted to bf16.
// rowp = &tile[rc][0]; elements 0..7 -> K = half*8 + e ; 8..15 -> K = 16 + half*8 + e.
__device__ __forceinline__ bf16x16 frag_from_lds(const float* rowp, int half) {
  bf16x16 v;
#pragma unroll
  for (int e = 0; e < 8; ++e) {
    v[e]     = (__bf16)rowp[half * 8 + e];
    v[e + 8] = (__bf16)rowp[16 + half * 8 + e];
  }
  return v;
}

// Direct-from-global A fragment (masked K tail only).
template <bool TAIL>
__device__ __forceinline__ bf16x16 load_a_frag(const float* __restrict__ A, long ld,
                                               int m0, int k0, int rc, int half, int K) {
  bf16x16 v;
  const float* p = A + (long)(m0 + rc) * ld + k0 + half * 8;
#pragma unroll
  for (int e = 0; e < 8; ++e) {
    float f0 = 0.0f, f1 = 0.0f;
    if (!TAIL || (k0 + half * 8 + e) < K)      f0 = p[e];
    if (!TAIL || (k0 + 16 + half * 8 + e) < K) f1 = p[e + 16];
    v[e]     = (__bf16)f0;
    v[e + 8] = (__bf16)f1;
  }
  return v;
}

// B fragment: 32x16 tile B[k][n] = W[n0+n][k0+k], W row-major (N,K), ld=K.
// Lane holds column (lane&15); elements 0..7 -> K = k0 + half*8 + e ; 8..15 -> K = k0 + 16 + half*8 + (e-8).
template <bool TAIL>
__device__ __forceinline__ bf16x16 load_b_frag(const float* __restrict__ W, long ld,
                                               int n0, int k0, int rc, int half, int K) {
  bf16x16 v;
  const float* p = W + (long)(n0 + rc) * ld + k0 + half * 8;
#pragma unroll
  for (int e = 0; e < 16; ++e) {
    const int kk = (e & 7) + ((e >> 3) * 16);
    float f = 0.0f;
    if (!TAIL || (k0 + half * 8 + kk) < K) f = p[kk];
    v[e] = (__bf16)f;
  }
  return v;
}

__device__ __forceinline__ float sigmoidf(float x) { return 1.0f / (1.0f + __expf(-x)); }

// ---------------- Generic GEMM: C[M,N] = A[M,K] @ W[N,K]^T + bias[N] ----------------
// block = 128 threads (4 waves); block computes a 16(M) x 128(N) strip.
// A tile staged once per block via async global->LDS, double-buffered.
// grid = (M/16, N/128). M % 16 == 0, N % 128 == 0 (true for all call sites).
__global__ __launch_bounds__(128) void gemm_bias_wmma(
    const float* __restrict__ A, const float* __restrict__ W,
    const float* __restrict__ bias, float* __restrict__ C,
    int M, int N, int K) {
  __shared__ float As[2][16][32];
  const int tid  = threadIdx.x;
  const int wave = tid >> 5;
  const int lane = tid & 31;
  const int half = lane >> 4, rc = lane & 15;
  const int m0 = blockIdx.x * 16;
  const int n0 = blockIdx.y * 128 + wave * 32;

  f32x8 acc0 = {}, acc1 = {};
  const int Kfull = K & ~31;
  int cur = 0;
  if (Kfull > 0) stage_a_async(&As[0][0][0], A, K, m0, 0, tid);
  for (int k0 = 0; k0 < Kfull; k0 += 32) {
    const bool more = (k0 + 32 < Kfull);
    if (more) {
      stage_a_async(&As[cur ^ 1][0][0], A, K, m0, k0 + 32, tid);
      __builtin_prefetch(W + (long)(n0 + rc) * K + k0 + 32 + half * 16, 0, 0);
      wait_async_le1();   // oldest (current) tile complete; next still in flight
    } else {
      wait_async_le0();
    }
    __syncthreads();
    bf16x16 a  = frag_from_lds(&As[cur][rc][0], half);
    bf16x16 b0 = load_b_frag<false>(W, K, n0,      k0, rc, half, K);
    bf16x16 b1 = load_b_frag<false>(W, K, n0 + 16, k0, rc, half, K);
    acc0 = wmma_bf16(a, b0, acc0);
    acc1 = wmma_bf16(a, b1, acc1);
    __syncthreads();      // all waves done reading before buffer reuse
    cur ^= 1;
  }
  if (Kfull < K) {  // K tail (GLOVE = 300): masked direct loads
    const int k0 = Kfull;
    bf16x16 a  = load_a_frag<true>(A, K, m0, k0, rc, half, K);
    bf16x16 b0 = load_b_frag<true>(W, K, n0,      k0, rc, half, K);
    bf16x16 b1 = load_b_frag<true>(W, K, n0 + 16, k0, rc, half, K);
    acc0 = wmma_bf16(a, b0, acc0);
    acc1 = wmma_bf16(a, b1, acc1);
  }
  const float bn0 = bias[n0 + rc];
  const float bn1 = bias[n0 + 16 + rc];
#pragma unroll
  for (int r = 0; r < 8; ++r) {
    const int m = m0 + r + half * 8;
    C[(long)m * N + n0 + rc]      = acc0[r] + bn0;
    C[(long)m * N + n0 + 16 + rc] = acc1[r] + bn1;
  }
}

// ---------------- Fused LSTM gate GEMM for one timestep ----------------
// g[b, n] = bih[n] + bhh[n] + x_t[b,:] @ Wih[n,:]^T + h[b,:] @ Whh[n,:]^T
// x_t rows live at xseq[(b*T + t)*XD ...] -> base = xseq + t*XD, row stride T*XD.
// grid = (batch/16, 4H/128), block = 128. XD % 32 == 0, H % 32 == 0.
__global__ __launch_bounds__(128) void lstm_gate_gemm(
    const float* __restrict__ xseq, const float* __restrict__ hprev,
    const float* __restrict__ Wih, const float* __restrict__ Whh,
    const float* __restrict__ bih, const float* __restrict__ bhh,
    float* __restrict__ g, int T, int t, int XD, int H, int N4H) {
  __shared__ float As[2][16][32];
  const int tid  = threadIdx.x;
  const int wave = tid >> 5;
  const int lane = tid & 31;
  const int half = lane >> 4, rc = lane & 15;
  const int m0 = blockIdx.x * 16;
  const int n0 = blockIdx.y * 128 + wave * 32;

  f32x8 acc0 = {}, acc1 = {};

  // phase 1: x_t @ Wih^T
  {
    const float* xbase = xseq + (long)t * XD;
    const long ld = (long)T * XD;
    int cur = 0;
    stage_a_async(&As[0][0][0], xbase, ld, m0, 0, tid);
    for (int k0 = 0; k0 < XD; k0 += 32) {
      const bool more = (k0 + 32 < XD);
      if (more) { stage_a_async(&As[cur ^ 1][0][0], xbase, ld, m0, k0 + 32, tid); wait_async_le1(); }
      else      { wait_async_le0(); }
      __syncthreads();
      bf16x16 a  = frag_from_lds(&As[cur][rc][0], half);
      bf16x16 b0 = load_b_frag<false>(Wih, XD, n0,      k0, rc, half, XD);
      bf16x16 b1 = load_b_frag<false>(Wih, XD, n0 + 16, k0, rc, half, XD);
      acc0 = wmma_bf16(a, b0, acc0);
      acc1 = wmma_bf16(a, b1, acc1);
      __syncthreads();
      cur ^= 1;
    }
  }
  // phase 2: h @ Whh^T
  {
    int cur = 0;
    stage_a_async(&As[0][0][0], hprev, H, m0, 0, tid);
    for (int k0 = 0; k0 < H; k0 += 32) {
      const bool more = (k0 + 32 < H);
      if (more) { stage_a_async(&As[cur ^ 1][0][0], hprev, H, m0, k0 + 32, tid); wait_async_le1(); }
      else      { wait_async_le0(); }
      __syncthreads();
      bf16x16 a  = frag_from_lds(&As[cur][rc][0], half);
      bf16x16 b0 = load_b_frag<false>(Whh, H, n0,      k0, rc, half, H);
      bf16x16 b1 = load_b_frag<false>(Whh, H, n0 + 16, k0, rc, half, H);
      acc0 = wmma_bf16(a, b0, acc0);
      acc1 = wmma_bf16(a, b1, acc1);
      __syncthreads();
      cur ^= 1;
    }
  }

  const float bn0 = bih[n0 + rc]      + bhh[n0 + rc];
  const float bn1 = bih[n0 + 16 + rc] + bhh[n0 + 16 + rc];
#pragma unroll
  for (int r = 0; r < 8; ++r) {
    const int m = m0 + r + half * 8;
    g[(long)m * N4H + n0 + rc]      = acc0[r] + bn0;
    g[(long)m * N4H + n0 + 16 + rc] = acc1[r] + bn1;
  }
}

// ---------------- Elementwise LSTM cell update ----------------
__global__ void lstm_cell_kernel(const float* __restrict__ g, float* __restrict__ c,
                                 float* __restrict__ h, float* __restrict__ hseq,
                                 int batch, int T, int t, int H) {
  const int i = blockIdx.x * blockDim.x + threadIdx.x;
  if (i >= batch * H) return;
  const int b = i / H, j = i - b * H;
  const long gb = (long)b * 4 * H;
  const float gi = g[gb + j];
  const float gf = g[gb + H + j];
  const float gg = g[gb + 2 * H + j];
  const float go = g[gb + 3 * H + j];
  const float cc = sigmoidf(gf) * c[i] + sigmoidf(gi) * tanhf(gg);
  const float hh = sigmoidf(go) * tanhf(cc);
  c[i] = cc;
  h[i] = hh;
  hseq[((long)b * T + t) * H + j] = hh;
}

// ---------------- Attention + masked softmax + mean over v ----------------
// One block per (bc, r): grid = 16*32, block = 256 (8 waves).
#define T_C 20
#define A_DIM 256
#define H_DIM 512
__global__ __launch_bounds__(256) void attention_kernel(
    const float* __restrict__ v_lin,   // (16*32*32, 256), row = (bc*32+r)*32+v
    const float* __restrict__ c_lin,   // (16*20, 256)
    const float* __restrict__ cap_h,   // (16*20, 512)
    const int*   __restrict__ cap_len, // (16,)
    const float* __restrict__ W_att,   // (256,)
    const float* __restrict__ b_att,   // (1,)
    float* __restrict__ out_cap) {     // (16*32, 512)
  __shared__ float cls[T_C][A_DIM];    // 20 KB
  __shared__ float chs[T_C][H_DIM];    // 40 KB
  __shared__ float score_s[T_C];
  __shared__ float w_s[T_C];

  const int bc = blockIdx.x >> 5;
  const int r  = blockIdx.x & 31;
  const int tid = threadIdx.x;
  const int wave = tid >> 5, lane = tid & 31;

  for (int idx = tid; idx < T_C * A_DIM; idx += 256)
    cls[idx / A_DIM][idx % A_DIM] = c_lin[(long)bc * T_C * A_DIM + idx];
  for (int idx = tid; idx < T_C * H_DIM; idx += 256)
    chs[idx / H_DIM][idx % H_DIM] = cap_h[(long)bc * T_C * H_DIM + idx];
  int len = cap_len[bc];
  if (len < 1) len = 1;
  if (len > T_C) len = T_C;
  __syncthreads();

  float wa[8];
#pragma unroll
  for (int q = 0; q < 8; ++q) wa[q] = W_att[lane + 32 * q];
  const float batt = b_att[0];

  float acc0 = 0.0f, acc1 = 0.0f;
  for (int v = 0; v < 32; ++v) {
    const float* vrow = v_lin + ((long)(bc * 32 + r) * 32 + v) * A_DIM;
    float vl[8];
#pragma unroll
    for (int q = 0; q < 8; ++q) vl[q] = vrow[lane + 32 * q];

    // waves split the 20 timesteps; each wave reduces a 256-dim tanh-dot
    for (int t = wave; t < T_C; t += 8) {
      float s = 0.0f;
#pragma unroll
      for (int q = 0; q < 8; ++q)
        s += tanhf(vl[q] + cls[t][lane + 32 * q]) * wa[q];
      for (int off = 16; off > 0; off >>= 1) s += __shfl_xor(s, off, 32);
      if (lane == 0) score_s[t] = s + batt;
    }
    __syncthreads();

    if (tid == 0) {  // masked softmax over the (tiny) t axis
      float mx = -1e30f;
      for (int t = 0; t < len; ++t) mx = fmaxf(mx, score_s[t]);
      float sum = 0.0f;
      for (int t = 0; t < len; ++t) { float e = __expf(score_s[t] - mx); w_s[t] = e; sum += e; }
      const float inv = 1.0f / sum;
      for (int t = 0; t < len; ++t) w_s[t] *= inv;
      for (int t = len; t < T_C; ++t) w_s[t] = 0.0f;
    }
    __syncthreads();

    float a0 = 0.0f, a1 = 0.0f;
    for (int t = 0; t < T_C; ++t) {
      const float wt = w_s[t];
      a0 += wt * chs[t][tid];
      a1 += wt * chs[t][tid + 256];
    }
    acc0 += a0;
    acc1 += a1;
    __syncthreads();
  }

  float* o = out_cap + (long)(bc * 32 + r) * H_DIM;
  o[tid]       = acc0 * (1.0f / 32.0f);
  o[tid + 256] = acc1 * (1.0f / 32.0f);
}

// ---------------- Mean over TV of vid_h ----------------
__global__ void vid_mean_kernel(const float* __restrict__ vid_h, float* __restrict__ out,
                                int Bv, int T, int Hd) {
  const int i = blockIdx.x * blockDim.x + threadIdx.x;
  if (i >= Bv * Hd) return;
  const int b = i / Hd, h = i - b * Hd;
  float s = 0.0f;
  for (int t = 0; t < T; ++t) s += vid_h[((long)b * T + t) * Hd + h];
  out[i] = s * (1.0f / (float)T);
}

// ---------------- Host launch ----------------
extern "C" void kernel_launch(void* const* d_in, const int* in_sizes, int n_in,
                              void* d_out, int out_size, void* d_ws, size_t ws_size,
                              hipStream_t stream) {
  const float* video_fc  = (const float*)d_in[0];   // (512, 32, 2048)
  const float* video_cap = (const float*)d_in[1];   // (16, 20, 300)
  const int*   cap_len   = (const int*)  d_in[2];   // (16,)
  const float* W_vemb = (const float*)d_in[3];
  const float* b_vemb = (const float*)d_in[4];
  const float* W_wemb = (const float*)d_in[5];
  const float* b_wemb = (const float*)d_in[6];
  const float* Wih_v  = (const float*)d_in[7];
  const float* Whh_v  = (const float*)d_in[8];
  const float* bih_v  = (const float*)d_in[9];
  const float* bhh_v  = (const float*)d_in[10];
  const float* Wih_c  = (const float*)d_in[11];
  const float* Whh_c  = (const float*)d_in[12];
  const float* bih_c  = (const float*)d_in[13];
  const float* bhh_c  = (const float*)d_in[14];
  const float* W_vid  = (const float*)d_in[15];
  const float* b_vid  = (const float*)d_in[16];
  const float* W_sen  = (const float*)d_in[17];
  const float* b_sen  = (const float*)d_in[18];
  const float* W_att  = (const float*)d_in[19];
  const float* b_att  = (const float*)d_in[20];

  // ---- workspace carve (floats) ----
  float* p = (float*)d_ws;
  float* vid_x  = p; p += (long)16384 * 512;   // (B*TV, VEMB)
  float* vid_hs = p; p += (long)16384 * 512;   // (B*TV, H)
  float* v_lin  = p; p += (long)16384 * 256;   // (B*TV, A)
  float* g_v    = p; p += (long)512 * 2048;    // (B, 4H)
  float* h_v    = p; p += (long)512 * 512;
  float* c_v    = p; p += (long)512 * 512;
  float* cap_x  = p; p += (long)320 * 512;     // (BC*TC, WEMB)
  float* cap_hs = p; p += (long)320 * 512;     // (BC*TC, H)
  float* c_lin  = p; p += (long)320 * 256;     // (BC*TC, A)
  float* g_c    = p; p += (long)16 * 2048;
  float* h_c    = p; p += (long)16 * 512;
  float* c_c    = p; p += (long)16 * 512;

  const dim3 blk(128);

  // 1) Embeddings (big WMMA GEMMs)
  gemm_bias_wmma<<<dim3(16384 / 16, 512 / 128), blk, 0, stream>>>(
      video_fc, W_vemb, b_vemb, vid_x, 16384, 512, 2048);
  gemm_bias_wmma<<<dim3(320 / 16, 512 / 128), blk, 0, stream>>>(
      video_cap, W_wemb, b_wemb, cap_x, 320, 512, 300);

  // zero LSTM state
  hipMemsetAsync(h_v, 0, (size_t)512 * 512 * 4, stream);
  hipMemsetAsync(c_v, 0, (size_t)512 * 512 * 4, stream);
  hipMemsetAsync(h_c, 0, (size_t)16 * 512 * 4, stream);
  hipMemsetAsync(c_c, 0, (size_t)16 * 512 * 4, stream);

  // 2) Caption LSTM: 20 steps, batch 16
  for (int t = 0; t < 20; ++t) {
    lstm_gate_gemm<<<dim3(16 / 16, 2048 / 128), blk, 0, stream>>>(
        cap_x, h_c, Wih_c, Whh_c, bih_c, bhh_c, g_c, 20, t, 512, 512, 2048);
    lstm_cell_kernel<<<(16 * 512 + 255) / 256, 256, 0, stream>>>(
        g_c, c_c, h_c, cap_hs, 16, 20, t, 512);
  }

  // 3) Video LSTM: 32 steps, batch 512
  for (int t = 0; t < 32; ++t) {
    lstm_gate_gemm<<<dim3(512 / 16, 2048 / 128), blk, 0, stream>>>(
        vid_x, h_v, Wih_v, Whh_v, bih_v, bhh_v, g_v, 32, t, 512, 512, 2048);
    lstm_cell_kernel<<<(512 * 512 + 255) / 256, 256, 0, stream>>>(
        g_v, c_v, h_v, vid_hs, 512, 32, t, 512);
  }

  // 4) Attention projections
  gemm_bias_wmma<<<dim3(320 / 16, 256 / 128), blk, 0, stream>>>(
      cap_hs, W_sen, b_sen, c_lin, 320, 256, 512);
  gemm_bias_wmma<<<dim3(16384 / 16, 256 / 128), blk, 0, stream>>>(
      vid_hs, W_vid, b_vid, v_lin, 16384, 256, 512);

  // 5) Outputs
  float* out_vid = (float*)d_out;                 // (16*32, 512)
  float* out_cap = (float*)d_out + 16 * 32 * 512; // (16*32, 512)
  vid_mean_kernel<<<(512 * 512 + 255) / 256, 256, 0, stream>>>(
      vid_hs, out_vid, 512, 32, 512);
  attention_kernel<<<16 * 32, 256, 0, stream>>>(
      v_lin, c_lin, cap_hs, cap_len, W_att, b_att, out_cap);
}
